// GNN_1_395136991890
// MI455X (gfx1250) — matrix operand order repaired and checked by reading, not compiled, not run on hardware
//
#include <hip/hip_runtime.h>
#include <hip/hip_bf16.h>

typedef __bf16 v16bf __attribute__((ext_vector_type(16)));
typedef float  v8f   __attribute__((ext_vector_type(8)));

#define HID 64
#define SLOPE 0.01f
#define BN_EPS 1e-5f

__device__ __forceinline__ v8f wmma_bf16(v16bf a, v16bf b, v8f c) {
  // (neg_a, A, neg_b, B, c_mod, C, reuse_a, reuse_b)
  return __builtin_amdgcn_wmma_f32_16x16x32_bf16(false, a, false, b, (short)0, c,
                                                 false, false);
}

// Load one 16x32 bf16 A-fragment from a row of a [*,64] fp32 matrix,
// optionally adding a second matrix (GIN: h + agg) on the fly.
// ISA 16-bit A layout: lanes 0-15 -> K={koff..koff+7, koff+16..koff+23},
// lanes 16-31 use koff+8 (handled by caller via koff = kbase + half*8).
__device__ __forceinline__ v16bf load_a_frag(const float* __restrict__ row,
                                             const float* __restrict__ row2,
                                             int koff) {
  v16bf a;
#pragma unroll
  for (int i = 0; i < 8; ++i) {
    float x0 = row[koff + i];
    float x1 = row[koff + 16 + i];
    if (row2) { x0 += row2[koff + i]; x1 += row2[koff + 16 + i]; }
    a[i]     = (__bf16)x0;
    a[i + 8] = (__bf16)x1;
  }
  return a;
}

// ---------------------------------------------------------------------------
// Pack a 64x64 fp32 weight (row-major, W[k][n]) into 8 WMMA B-fragments
// (bf16, ISA 32x16 B layout: lanes 0-15 hold K={0..7,16..23} of column n=lane,
// lanes 16-31 hold K={8..15,24..31}). Fragment f = ktile*4 + ntile, 512 bf16 each.
// ---------------------------------------------------------------------------
__global__ void pack_b_kernel(const float* __restrict__ W, __bf16* __restrict__ out) {
  int idx  = blockIdx.x * blockDim.x + threadIdx.x;   // 0..4095
  int f    = idx >> 9;
  int rem  = idx & 511;
  int lane = rem >> 4;
  int i    = rem & 15;
  int t = f >> 2, j = f & 3;
  int n = j * 16 + (lane & 15);
  int k = t * 32 + ((lane & 16) ? 8 : 0) + (i & 7) + ((i & 8) ? 16 : 0);
  out[idx] = (__bf16)W[k * HID + n];
}

__global__ void zero_kernel(float* __restrict__ p, long long n) {
  long long i = blockIdx.x * (long long)blockDim.x + threadIdx.x;
  long long stride = (long long)gridDim.x * blockDim.x;
  for (; i < n; i += stride) p[i] = 0.0f;
}

// ---------------------------------------------------------------------------
// out[row] = in[row] @ W + bias    (pre_mp, no activation)
// One wave per 16-row tile, 8 waves / block.
// ---------------------------------------------------------------------------
__global__ __launch_bounds__(256)
void gemm_bias_kernel(const float* __restrict__ in, const __bf16* __restrict__ pw,
                      const float* __restrict__ bias, float* __restrict__ out,
                      int nrows) {
  int wave = threadIdx.x >> 5, lane = threadIdx.x & 31;
  int half = lane >> 4, r = lane & 15;
  int rowTile = blockIdx.x * 128 + wave * 16;
  int row  = rowTile + r;
  int rowc = row < nrows ? row : nrows - 1;
  const float* rb = in + (size_t)rowc * HID;

  v16bf a0 = load_a_frag(rb, nullptr, half * 8);
  v16bf a1 = load_a_frag(rb, nullptr, 32 + half * 8);

  v8f acc[4];
#pragma unroll
  for (int j = 0; j < 4; ++j) {
    float bv = bias[j * 16 + r];
#pragma unroll
    for (int v = 0; v < 8; ++v) acc[j][v] = bv;
  }
#pragma unroll
  for (int j = 0; j < 4; ++j) {
    v16bf b0 = *(const v16bf*)(pw + (0 * 4 + j) * 512 + lane * 16);
    v16bf b1 = *(const v16bf*)(pw + (1 * 4 + j) * 512 + lane * 16);
    acc[j] = wmma_bf16(a0, b0, acc[j]);
    acc[j] = wmma_bf16(a1, b1, acc[j]);
  }
#pragma unroll
  for (int j = 0; j < 4; ++j) {
    int col = j * 16 + r;
#pragma unroll
    for (int v = 0; v < 8; ++v) {
      int m = half ? v + 8 : v;
      int orow = rowTile + m;
      if (orow < nrows) out[(size_t)orow * HID + col] = acc[j][v];
    }
  }
}

// ---------------------------------------------------------------------------
// Fused MLP: out = leaky(( in [+ in2] ) @ W1 + b1) @ W2 + b2
// GEMM1 -> LeakyReLU -> LDS relayout (D-layout to A-layout) -> GEMM2.
// ---------------------------------------------------------------------------
__global__ __launch_bounds__(256)
void mlp2_kernel(const float* __restrict__ in, const float* __restrict__ in2,
                 const __bf16* __restrict__ pw1, const float* __restrict__ b1,
                 const __bf16* __restrict__ pw2, const float* __restrict__ b2,
                 float* __restrict__ out, int nrows) {
  __shared__ __bf16 lds[128][72];  // 18 KB, padded stride

  int wave = threadIdx.x >> 5, lane = threadIdx.x & 31;
  int half = lane >> 4, r = lane & 15;
  int rowTile = blockIdx.x * 128 + wave * 16;
  int row  = rowTile + r;
  int rowc = row < nrows ? row : nrows - 1;
  const float* rb  = in + (size_t)rowc * HID;
  const float* rb2 = in2 ? in2 + (size_t)rowc * HID : nullptr;

  v16bf a0 = load_a_frag(rb, rb2, half * 8);
  v16bf a1 = load_a_frag(rb, rb2, 32 + half * 8);

  v8f acc[4];
#pragma unroll
  for (int j = 0; j < 4; ++j) {
    float bv = b1[j * 16 + r];
#pragma unroll
    for (int v = 0; v < 8; ++v) acc[j][v] = bv;
  }
#pragma unroll
  for (int j = 0; j < 4; ++j) {
    v16bf w0 = *(const v16bf*)(pw1 + (0 * 4 + j) * 512 + lane * 16);
    v16bf w1 = *(const v16bf*)(pw1 + (1 * 4 + j) * 512 + lane * 16);
    acc[j] = wmma_bf16(a0, w0, acc[j]);
    acc[j] = wmma_bf16(a1, w1, acc[j]);
  }

  // LeakyReLU + stash in LDS (D layout -> row-major bf16)
  int lbase = wave * 16;
#pragma unroll
  for (int j = 0; j < 4; ++j) {
    int col = j * 16 + r;
#pragma unroll
    for (int v = 0; v < 8; ++v) {
      float x = acc[j][v];
      x = x > 0.0f ? x : SLOPE * x;
      int m = half ? v + 8 : v;
      lds[lbase + m][col] = (__bf16)x;
    }
  }
  __syncthreads();

  // Reload as A fragments (already bf16)
  const __bf16* lr = &lds[lbase + r][0];
  v16bf c0, c1;
#pragma unroll
  for (int i = 0; i < 8; ++i) {
    c0[i]     = lr[half * 8 + i];
    c0[i + 8] = lr[half * 8 + 16 + i];
    c1[i]     = lr[32 + half * 8 + i];
    c1[i + 8] = lr[32 + half * 8 + 16 + i];
  }

  v8f acc2[4];
#pragma unroll
  for (int j = 0; j < 4; ++j) {
    float bv = b2[j * 16 + r];
#pragma unroll
    for (int v = 0; v < 8; ++v) acc2[j][v] = bv;
  }
#pragma unroll
  for (int j = 0; j < 4; ++j) {
    v16bf w0 = *(const v16bf*)(pw2 + (0 * 4 + j) * 512 + lane * 16);
    v16bf w1 = *(const v16bf*)(pw2 + (1 * 4 + j) * 512 + lane * 16);
    acc2[j] = wmma_bf16(c0, w0, acc2[j]);
    acc2[j] = wmma_bf16(c1, w1, acc2[j]);
  }
#pragma unroll
  for (int j = 0; j < 4; ++j) {
    int col = j * 16 + r;
#pragma unroll
    for (int v = 0; v < 8; ++v) {
      int m = half ? v + 8 : v;
      int orow = rowTile + m;
      if (orow < nrows) out[(size_t)orow * HID + col] = acc2[j][v];
    }
  }
}

// ---------------------------------------------------------------------------
// GIN aggregation: agg[dst] += h[src]  (16 lanes per edge, float4 per lane)
// ---------------------------------------------------------------------------
__global__ void scatter_kernel(const float* __restrict__ h,
                               const long long* __restrict__ src,
                               const long long* __restrict__ dst,
                               float* __restrict__ agg, long long E) {
  long long tid = blockIdx.x * (long long)blockDim.x + threadIdx.x;
  long long e = tid >> 4;
  if (e >= E) return;
  int c = (int)(tid & 15) * 4;
  long long s = src[e], d = dst[e];
  const float4 v = *(const float4*)(h + s * HID + c);
  float* p = agg + d * HID + c;
  atomicAdd(p + 0, v.x);
  atomicAdd(p + 1, v.y);
  atomicAdd(p + 2, v.z);
  atomicAdd(p + 3, v.w);
}

// ---------------------------------------------------------------------------
// BatchNorm stats: per-column sum / sumsq (64 columns), block-local LDS
// reduce then a handful of global f32 atomics.
// ---------------------------------------------------------------------------
__global__ __launch_bounds__(256)
void bn_stats_kernel(const float* __restrict__ h, float* __restrict__ sums,
                     float* __restrict__ sumsq, int nrows) {
  __shared__ float sS[4][64], sQ[4][64];
  int col = threadIdx.x & 63;
  int sub = threadIdx.x >> 6;
  float s = 0.0f, q = 0.0f;
  long long stride = (long long)gridDim.x * 4;
  for (long long rowb = blockIdx.x * 4 + sub; rowb < nrows; rowb += stride) {
    float x = h[rowb * HID + col];
    s += x;
    q += x * x;
  }
  sS[sub][col] = s;
  sQ[sub][col] = q;
  __syncthreads();
  if (sub == 0) {
    float ts = sS[0][col] + sS[1][col] + sS[2][col] + sS[3][col];
    float tq = sQ[0][col] + sQ[1][col] + sQ[2][col] + sQ[3][col];
    atomicAdd(&sums[col], ts);
    atomicAdd(&sumsq[col], tq);
  }
}

__global__ void bn_apply_kernel(const float* __restrict__ h,
                                const float* __restrict__ sums,
                                const float* __restrict__ sumsq,
                                const float* __restrict__ g,
                                const float* __restrict__ b,
                                float* __restrict__ out, int nrows) {
  long long idx = blockIdx.x * (long long)blockDim.x + threadIdx.x;
  long long n = (long long)nrows * HID;
  if (idx >= n) return;
  int col = (int)(idx & 63);
  float inv_n = 1.0f / (float)nrows;
  float mu  = sums[col] * inv_n;
  float var = sumsq[col] * inv_n - mu * mu;
  out[idx] = (h[idx] - mu) * rsqrtf(var + BN_EPS) * g[col] + b[col];
}

// ---------------------------------------------------------------------------
extern "C" void kernel_launch(void* const* d_in, const int* in_sizes, int n_in,
                              void* d_out, int out_size, void* d_ws, size_t ws_size,
                              hipStream_t stream) {
  const int N = in_sizes[0] / HID;           // 40000
  const long long E = in_sizes[1] / 2;       // 1280000

  const float* x           = (const float*)d_in[0];
  const long long* ei      = (const long long*)d_in[1];
  const float* W_pre = (const float*)d_in[2];
  const float* b_pre = (const float*)d_in[3];
  const float* c0_W1 = (const float*)d_in[4];
  const float* c0_b1 = (const float*)d_in[5];
  const float* c0_W2 = (const float*)d_in[6];
  const float* c0_b2 = (const float*)d_in[7];
  const float* bn_g  = (const float*)d_in[8];
  const float* bn_b  = (const float*)d_in[9];
  const float* c1_W1 = (const float*)d_in[10];
  const float* c1_b1 = (const float*)d_in[11];
  const float* c1_W2 = (const float*)d_in[12];
  const float* c1_b2 = (const float*)d_in[13];
  const float* Wp1   = (const float*)d_in[14];
  const float* bp1   = (const float*)d_in[15];
  const float* Wp2   = (const float*)d_in[16];
  const float* bp2   = (const float*)d_in[17];

  // Workspace layout
  char* ws = (char*)d_ws;
  __bf16* pw   = (__bf16*)ws;                         // 7*4096 bf16 = 57344 B
  float* stats = (float*)(ws + 57344);                // 128 floats
  float* bufA  = (float*)(ws + 57856);                // N*64
  float* bufB  = bufA + (size_t)N * HID;
  float* bufC  = bufB + (size_t)N * HID;

  // Pack all weight matrices into bf16 WMMA B-fragments (cheap, L1-resident)
  const float* Ws[7] = {W_pre, c0_W1, c0_W2, c1_W1, c1_W2, Wp1, Wp2};
  for (int m = 0; m < 7; ++m)
    pack_b_kernel<<<16, 256, 0, stream>>>(Ws[m], pw + m * 4096);

  dim3 gGemm((N + 127) / 128);
  long long nElem = (long long)N * HID;
  int scBlocks = (int)((E * 16 + 255) / 256);

  // pre_mp: h0 = x @ W_pre + b_pre        -> bufA
  gemm_bias_kernel<<<gGemm, 256, 0, stream>>>(x, pw + 0 * 4096, b_pre, bufA, N);

  // conv0: agg = segment_sum(h0[src], dst) -> bufB
  zero_kernel<<<1024, 256, 0, stream>>>(bufB, nElem);
  scatter_kernel<<<scBlocks, 256, 0, stream>>>(bufA, ei, ei + E, bufB, E);
  // h1 = MLP(h0 + agg)                     -> bufC
  mlp2_kernel<<<gGemm, 256, 0, stream>>>(bufA, bufB, pw + 1 * 4096, c0_b1,
                                         pw + 2 * 4096, c0_b2, bufC, N);

  // BatchNorm(h1)                          -> bufA
  zero_kernel<<<1, 128, 0, stream>>>(stats, 128);
  bn_stats_kernel<<<160, 256, 0, stream>>>(bufC, stats, stats + 64, N);
  bn_apply_kernel<<<(int)((nElem + 255) / 256), 256, 0, stream>>>(
      bufC, stats, stats + 64, bn_g, bn_b, bufA, N);

  // conv1
  zero_kernel<<<1024, 256, 0, stream>>>(bufB, nElem);
  scatter_kernel<<<scBlocks, 256, 0, stream>>>(bufA, ei, ei + E, bufB, E);
  mlp2_kernel<<<gGemm, 256, 0, stream>>>(bufA, bufB, pw + 3 * 4096, c1_b1,
                                         pw + 4 * 4096, c1_b2, bufC, N);

  // post_mp -> d_out  (flat [N,64] == [B, 5000*64])
  mlp2_kernel<<<gGemm, 256, 0, stream>>>(bufC, nullptr, pw + 5 * 4096, bp1,
                                         pw + 6 * 4096, bp2, (float*)d_out, N);
}
